// mGCN_87273735454839
// MI455X (gfx1250) — compile-verified, matrix operand order checked by reading, not run on hardware
//
#include <hip/hip_runtime.h>
#include <hip/hip_bf16.h>

// ---------------------------------------------------------------------------
// Problem constants (from the reference)
// ---------------------------------------------------------------------------
static constexpr int   NN    = 50000;
static constexpr int   EE    = 800000;
static constexpr int   VV    = 3;
static constexpr int   DIN   = 128;
static constexpr int   DH    = 128;
static constexpr int   DOUT  = 40;
static constexpr int   DOUTP = 48;          // DOUT padded to a multiple of 16
static constexpr int   DEMB  = VV * DH;     // 384
static constexpr float ALPHA = 0.5f;

typedef __attribute__((ext_vector_type(16))) __bf16 v16bf;
typedef __attribute__((ext_vector_type(8)))  __bf16 v8bf;
typedef __attribute__((ext_vector_type(8)))  float  v8f;

__device__ __forceinline__ float eluf(float x) {
    return x > 0.f ? x : (__expf(x) - 1.f);
}

// ---------------------------------------------------------------------------
// fp32 -> bf16 conversion
// ---------------------------------------------------------------------------
__global__ void f32_to_bf16_kernel(const float* __restrict__ in,
                                   __hip_bfloat16* __restrict__ out, long n) {
    long i = (long)blockIdx.x * blockDim.x + threadIdx.x;
    if (i < n) out[i] = __float2bfloat16(in[i]);
}

// C2w (40 x 384) -> padded (48 x 384) bf16, zero rows 40..47
__global__ void pad_c2w_kernel(const float* __restrict__ in,
                               __hip_bfloat16* __restrict__ out) {
    int i = blockIdx.x * blockDim.x + threadIdx.x;
    if (i >= DOUTP * DEMB) return;
    int r = i / DEMB, c = i % DEMB;
    out[i] = __float2bfloat16(r < DOUT ? in[r * DEMB + c] : 0.f);
}

// ---------------------------------------------------------------------------
// Attention: T[v][h][j] = sum_i W[v][h][i] * A[i][j]
// ---------------------------------------------------------------------------
__global__ void att_T_kernel(const float* __restrict__ W,
                             const float* __restrict__ A,
                             float* __restrict__ T) {
    int idx = blockIdx.x * blockDim.x + threadIdx.x;   // over VV*DH*DH
    if (idx >= VV * DH * DH) return;
    int vh = idx >> 7;
    int j  = idx & 127;
    const float* wrow = W + (long)vh * DH;
    float s = 0.f;
    #pragma unroll 4
    for (int i = 0; i < DH; ++i) s += wrow[i] * A[i * DH + j];
    T[idx] = s;
}

// Mraw[v][u] = sum_{h,j} T[v][h][j] * W[u][h][j]   (one block per (v,u))
__global__ void att_M_kernel(const float* __restrict__ T,
                             const float* __restrict__ W,
                             float* __restrict__ Mraw) {
    __shared__ float red[256];
    int b = blockIdx.x;
    int v = b / VV, u = b % VV;
    const float* Tv = T + (long)v * DH * DH;
    const float* Wu = W + (long)u * DH * DH;
    float s = 0.f;
    for (int i = threadIdx.x; i < DH * DH; i += blockDim.x) s += Tv[i] * Wu[i];
    red[threadIdx.x] = s;
    __syncthreads();
    for (int o = 128; o > 0; o >>= 1) {
        if ((int)threadIdx.x < o) red[threadIdx.x] += red[threadIdx.x + o];
        __syncthreads();
    }
    if (threadIdx.x == 0) Mraw[b] = red[0];
}

// 3x3 row softmax of (Mraw + DH*ab)
__global__ void att_softmax_kernel(const float* __restrict__ Mraw,
                                   const float* __restrict__ ab,
                                   float* __restrict__ att) {
    if (threadIdx.x != 0 || blockIdx.x != 0) return;
    float bias = (float)DH * ab[0];
    for (int v = 0; v < VV; ++v) {
        float m0 = Mraw[v * 3 + 0] + bias;
        float m1 = Mraw[v * 3 + 1] + bias;
        float m2 = Mraw[v * 3 + 2] + bias;
        float mx = fmaxf(m0, fmaxf(m1, m2));
        float e0 = __expf(m0 - mx), e1 = __expf(m1 - mx), e2 = __expf(m2 - mx);
        float s  = e0 + e1 + e2;
        att[v * 3 + 0] = e0 / s;
        att[v * 3 + 1] = e1 / s;
        att[v * 3 + 2] = e2 / s;
    }
}

// ---------------------------------------------------------------------------
// Degrees (shared by both layers: same edges)
// ---------------------------------------------------------------------------
__global__ void deg_init_kernel(float* __restrict__ deg, long n) {
    long i = (long)blockIdx.x * blockDim.x + threadIdx.x;
    if (i < n) deg[i] = 1.0f;                  // self loop
}

__global__ void deg_count_kernel(const int* __restrict__ ei,
                                 float* __restrict__ deg) {
    long i = (long)blockIdx.x * blockDim.x + threadIdx.x;  // over VV*EE
    if (i >= (long)VV * EE) return;
    int v = (int)(i / EE), e = (int)(i % EE);
    int d = ei[(long)v * 2 * EE + EE + e];     // dst
    atomicAdd(&deg[(long)v * NN + d], 1.0f);
}

__global__ void deg_rsqrt_kernel(float* __restrict__ deg, long n) {
    long i = (long)blockIdx.x * blockDim.x + threadIdx.x;
    if (i < n) deg[i] = rsqrtf(deg[i]);        // deg buffer now holds dis
}

// ---------------------------------------------------------------------------
// Batched bf16 WMMA GEMM:  C[z][m][n] = sum_k A[m][k] * B[z][n][k]  (+bias, elu)
// A: M x K row-major bf16 (shared across z), B: ceil16(Nout) x K row-major bf16.
// One wave -> one 16x16 output tile. M must be a multiple of 16; K of 32.
// ---------------------------------------------------------------------------
__device__ __forceinline__ v16bf load_frag(const __bf16* p0, const __bf16* p1) {
    v8bf lo = *(const v8bf*)p0;
    v8bf hi = *(const v8bf*)p1;
    v16bf r;
    #pragma unroll
    for (int i = 0; i < 8; ++i) { r[i] = lo[i]; r[i + 8] = hi[i]; }
    return r;
}

__global__ void wmma_gemm_kernel(const __hip_bfloat16* __restrict__ Abf,
                                 const __hip_bfloat16* __restrict__ Bbf,
                                 const float* __restrict__ bias,
                                 float* __restrict__ Cf,
                                 __hip_bfloat16* __restrict__ Cb,
                                 int M, int Nout, int K,
                                 long strideB, long strideC, int elu_flag) {
    const int wavesPerBlock = blockDim.x >> 5;
    int wave = blockIdx.x * wavesPerBlock + ((int)threadIdx.x >> 5);
    int lane = threadIdx.x & 31;
    int tiles_n = (Nout + 15) >> 4;
    int tiles_m = M >> 4;
    if (wave >= tiles_m * tiles_n) return;     // wave-uniform: EXEC all-1 at WMMA
    int tm = wave / tiles_n;
    int tn = wave % tiles_n;
    int z  = blockIdx.z;

    const __bf16* A = (const __bf16*)Abf;
    const __bf16* B = (const __bf16*)Bbf + (long)z * strideB;

    int half = lane >> 4;      // 0: K low halves, 1: K high halves (per ISA layout)
    int l15  = lane & 15;
    const __bf16* arow = A + (long)(tm * 16 + l15) * K;
    const __bf16* brow = B + (long)(tn * 16 + l15) * K;

    v8f acc = {};
    for (int kb = 0; kb < K; kb += 32) {
        // elems 0..7 = K[kb + 8*half .. +7], elems 8..15 = K[kb+16+8*half .. +7]
        v16bf a = load_frag(arow + kb + half * 8, arow + kb + 16 + half * 8);
        v16bf b = load_frag(brow + kb + half * 8, brow + kb + 16 + half * 8);
        acc = __builtin_amdgcn_wmma_f32_16x16x32_bf16(
                  false, a, false, b, (short)0, acc, false, false);
    }

    int col = tn * 16 + l15;
    if (col < Nout) {
        float bv = bias ? bias[col] : 0.f;
        #pragma unroll
        for (int r = 0; r < 8; ++r) {
            int row  = tm * 16 + r + half * 8;   // C layout: VGPR r, lane half
            float vv = acc[r] + bv;
            if (elu_flag) vv = eluf(vv);
            long idx = (long)z * strideC + (long)row * Nout + col;
            if (Cf) Cf[idx] = vv;
            else    Cb[idx] = __float2bfloat16(vv);
        }
    }
}

// ---------------------------------------------------------------------------
// agg[v][n][h] = lin[v][n][h] * dis[v][n]^2   (self-loop term; full overwrite)
// ---------------------------------------------------------------------------
__global__ void agg_init_kernel(const float* __restrict__ lin,
                                const float* __restrict__ dis,
                                float* __restrict__ agg) {
    long idx = (long)blockIdx.x * blockDim.x + threadIdx.x;   // VV*NN*DH
    if (idx >= (long)VV * NN * DH) return;
    long vn = idx >> 7;
    float d = dis[vn];
    agg[idx] = lin[idx] * d * d;
}

// One wave per edge; lane j handles h = lane, lane+32, lane+64, lane+96.
__global__ void edge_scatter_kernel(const int* __restrict__ ei,
                                    const float* __restrict__ dis,
                                    const float* __restrict__ lin,
                                    float* __restrict__ agg) {
    int wave = blockIdx.x * (blockDim.x >> 5) + ((int)threadIdx.x >> 5);
    int lane = threadIdx.x & 31;
    if (wave >= VV * EE) return;
    int v = wave / EE, e = wave % EE;
    int s = ei[(long)v * 2 * EE + e];
    int d = ei[(long)v * 2 * EE + EE + e];
    float norm = dis[(long)v * NN + s] * dis[(long)v * NN + d];
    const float* srcp = lin + ((long)v * NN + s) * DH;
    float*       dstp = agg + ((long)v * NN + d) * DH;
    #pragma unroll
    for (int j = 0; j < 4; ++j) {
        int h = lane + 32 * j;
        atomicAdd(&dstp[h], srcp[h] * norm);
    }
}

// ---------------------------------------------------------------------------
// cross + transpose into emb layout (N, V*DH) as bf16:
//   hmulti = agg + b   (elu'd for layer 1)
//   emb[n][v*DH+h] = 1-ALPHA + hmulti + elu(ALPHA * sum_u att[v][u]*lin[u][n][h])
// ---------------------------------------------------------------------------
__global__ void cross_kernel(const float* __restrict__ agg,
                             const float* __restrict__ lin,
                             const float* __restrict__ bias,
                             const float* __restrict__ att,
                             __hip_bfloat16* __restrict__ emb,
                             int elu_h) {
    long idx = (long)blockIdx.x * blockDim.x + threadIdx.x;   // NN*DEMB
    if (idx >= (long)NN * DEMB) return;
    long n = idx / DEMB;
    int  k = (int)(idx % DEMB);
    int  v = k >> 7;
    int  h = k & 127;
    float hv = agg[((long)v * NN + n) * DH + h] + bias[v * DH + h];
    if (elu_h) hv = eluf(hv);
    float c = ALPHA * (att[v * 3 + 0] * lin[((long)0 * NN + n) * DH + h] +
                       att[v * 3 + 1] * lin[((long)1 * NN + n) * DH + h] +
                       att[v * 3 + 2] * lin[((long)2 * NN + n) * DH + h]);
    emb[idx] = __float2bfloat16(1.0f - ALPHA + hv + eluf(c));
}

// ---------------------------------------------------------------------------
// log_softmax over DOUT=40 classes; one wave per row.
// ---------------------------------------------------------------------------
__global__ void logsoftmax_kernel(const float* __restrict__ in,
                                  float* __restrict__ out) {
    int n    = blockIdx.x * (blockDim.x >> 5) + ((int)threadIdx.x >> 5);
    int lane = threadIdx.x & 31;
    if (n >= NN) return;
    float v0 = (lane      < DOUT) ? in[(long)n * DOUT + lane]      : -3.4e38f;
    float v1 = (lane + 32 < DOUT) ? in[(long)n * DOUT + lane + 32] : -3.4e38f;
    float m = fmaxf(v0, v1);
    for (int o = 16; o > 0; o >>= 1) m = fmaxf(m, __shfl_xor(m, o, 32));
    float s = ((lane      < DOUT) ? __expf(v0 - m) : 0.f) +
              ((lane + 32 < DOUT) ? __expf(v1 - m) : 0.f);
    for (int o = 16; o > 0; o >>= 1) s += __shfl_xor(s, o, 32);
    float l = __logf(s);
    if (lane      < DOUT) out[(long)n * DOUT + lane]      = v0 - m - l;
    if (lane + 32 < DOUT) out[(long)n * DOUT + lane + 32] = v1 - m - l;
}

// ---------------------------------------------------------------------------
// Host side
// ---------------------------------------------------------------------------
extern "C" void kernel_launch(void* const* d_in, const int* in_sizes, int n_in,
                              void* d_out, int out_size, void* d_ws, size_t ws_size,
                              hipStream_t stream) {
    const float* x   = (const float*)d_in[0];
    const int*   ei  = (const int*)  d_in[1];
    const float* W1  = (const float*)d_in[2];
    const float* b1  = (const float*)d_in[3];
    const float* W2  = (const float*)d_in[4];
    const float* b2  = (const float*)d_in[5];
    const float* A1  = (const float*)d_in[6];
    const float* ab1 = (const float*)d_in[7];
    const float* A2  = (const float*)d_in[8];
    const float* ab2 = (const float*)d_in[9];
    const float* C1w = (const float*)d_in[10];
    const float* C1b = (const float*)d_in[11];
    const float* C2w = (const float*)d_in[12];
    const float* C2b = (const float*)d_in[13];
    float* out = (float*)d_out;

    // Workspace carve-up (256B aligned slabs)
    size_t off = 0;
    auto alloc = [&](size_t bytes) -> void* {
        void* p = (char*)d_ws + off;
        off = (off + bytes + 255) & ~(size_t)255;
        return p;
    };
    __hip_bfloat16* x_bf   = (__hip_bfloat16*)alloc((size_t)NN * DIN * 2);
    __hip_bfloat16* xc_bf  = (__hip_bfloat16*)alloc((size_t)NN * DH * 2);
    __hip_bfloat16* W1_bf  = (__hip_bfloat16*)alloc((size_t)VV * DH * DIN * 2);
    __hip_bfloat16* W2_bf  = (__hip_bfloat16*)alloc((size_t)VV * DH * DH * 2);
    __hip_bfloat16* C1w_bf = (__hip_bfloat16*)alloc((size_t)DH * DEMB * 2);
    __hip_bfloat16* C2w_bf = (__hip_bfloat16*)alloc((size_t)DOUTP * DEMB * 2);
    float* Tbuf   = (float*)alloc((size_t)VV * DH * DH * 4);
    float* Mraw   = (float*)alloc(9 * 4);
    float* att1   = (float*)alloc(9 * 4);
    float* att2   = (float*)alloc(9 * 4);
    float* dis    = (float*)alloc((size_t)VV * NN * 4);
    float* lin    = (float*)alloc((size_t)VV * NN * DH * 4);
    float* agg    = (float*)alloc((size_t)VV * NN * DH * 4);
    __hip_bfloat16* emb_bf = (__hip_bfloat16*)alloc((size_t)NN * DEMB * 2);
    float* outpre = (float*)alloc((size_t)NN * DOUT * 4);

    auto cdiv = [](long a, long b) { return (unsigned)((a + b - 1) / b); };

    // --- conversions -------------------------------------------------------
    f32_to_bf16_kernel<<<cdiv((long)NN * DIN, 256), 256, 0, stream>>>(x, x_bf, (long)NN * DIN);
    f32_to_bf16_kernel<<<cdiv((long)VV * DH * DIN, 256), 256, 0, stream>>>(W1, W1_bf, (long)VV * DH * DIN);
    f32_to_bf16_kernel<<<cdiv((long)VV * DH * DH, 256), 256, 0, stream>>>(W2, W2_bf, (long)VV * DH * DH);
    f32_to_bf16_kernel<<<cdiv((long)DH * DEMB, 256), 256, 0, stream>>>(C1w, C1w_bf, (long)DH * DEMB);
    pad_c2w_kernel<<<cdiv((long)DOUTP * DEMB, 256), 256, 0, stream>>>(C2w, C2w_bf);

    // --- attention matrices (3x3) -----------------------------------------
    att_T_kernel<<<cdiv(VV * DH * DH, 256), 256, 0, stream>>>(W1, A1, Tbuf);
    att_M_kernel<<<9, 256, 0, stream>>>(Tbuf, W1, Mraw);
    att_softmax_kernel<<<1, 32, 0, stream>>>(Mraw, ab1, att1);
    att_T_kernel<<<cdiv(VV * DH * DH, 256), 256, 0, stream>>>(W2, A2, Tbuf);
    att_M_kernel<<<9, 256, 0, stream>>>(Tbuf, W2, Mraw);
    att_softmax_kernel<<<1, 32, 0, stream>>>(Mraw, ab2, att2);

    // --- degrees / inverse sqrt (same for both layers) --------------------
    deg_init_kernel<<<cdiv((long)VV * NN, 256), 256, 0, stream>>>(dis, (long)VV * NN);
    deg_count_kernel<<<cdiv((long)VV * EE, 256), 256, 0, stream>>>(ei, dis);
    deg_rsqrt_kernel<<<cdiv((long)VV * NN, 256), 256, 0, stream>>>(dis, (long)VV * NN);

    // GEMM launch helper numbers: 8 waves / 256-thread block, 1 tile per wave
    const int tilesM = NN / 16;                       // 3125

    // --- layer 1: lin1[v] = x @ W1[v]^T  (bf16 WMMA, fp32 out) ------------
    {
        int tiles = tilesM * (DH / 16);
        dim3 g(cdiv(tiles, 8), 1, VV);
        wmma_gemm_kernel<<<g, 256, 0, stream>>>(x_bf, W1_bf, nullptr, lin, nullptr,
                                                NN, DH, DIN,
                                                (long)DH * DIN, (long)NN * DH, 0);
    }
    agg_init_kernel<<<cdiv((long)VV * NN * DH, 256), 256, 0, stream>>>(lin, dis, agg);
    edge_scatter_kernel<<<cdiv((long)VV * EE, 8), 256, 0, stream>>>(ei, dis, lin, agg);
    cross_kernel<<<cdiv((long)NN * DEMB, 256), 256, 0, stream>>>(agg, lin, b1, att1, emb_bf, 1);

    // --- combine1: xc = elu(emb @ C1w^T + C1b) -> bf16 --------------------
    {
        int tiles = tilesM * (DH / 16);
        dim3 g(cdiv(tiles, 8), 1, 1);
        wmma_gemm_kernel<<<g, 256, 0, stream>>>(emb_bf, C1w_bf, C1b, nullptr, xc_bf,
                                                NN, DH, DEMB, 0, 0, 1);
    }

    // --- layer 2: lin2[v] = xc @ W2[v]^T ----------------------------------
    {
        int tiles = tilesM * (DH / 16);
        dim3 g(cdiv(tiles, 8), 1, VV);
        wmma_gemm_kernel<<<g, 256, 0, stream>>>(xc_bf, W2_bf, nullptr, lin, nullptr,
                                                NN, DH, DH,
                                                (long)DH * DH, (long)NN * DH, 0);
    }
    agg_init_kernel<<<cdiv((long)VV * NN * DH, 256), 256, 0, stream>>>(lin, dis, agg);
    edge_scatter_kernel<<<cdiv((long)VV * EE, 8), 256, 0, stream>>>(ei, dis, lin, agg);
    cross_kernel<<<cdiv((long)NN * DEMB, 256), 256, 0, stream>>>(agg, lin, b2, att2, emb_bf, 0);

    // --- combine2: outpre = elu(emb @ C2w^T + C2b) ------------------------
    {
        int tiles = tilesM * (DOUTP / 16);
        dim3 g(cdiv(tiles, 8), 1, 1);
        wmma_gemm_kernel<<<g, 256, 0, stream>>>(emb_bf, C2w_bf, C2b, outpre, nullptr,
                                                NN, DOUT, DEMB, 0, 0, 1);
    }

    // --- log_softmax over classes -----------------------------------------
    logsoftmax_kernel<<<cdiv(NN, 8), 256, 0, stream>>>(outpre, out);
}
